// GraphSAGE_33801392619948
// MI455X (gfx1250) — matrix operand order repaired and checked by reading, not compiled, not run on hardware
//
#include <hip/hip_runtime.h>
#include <hip/hip_bf16.h>
#include <stdint.h>

#define DH 128

typedef __attribute__((ext_vector_type(16))) __bf16 v16bf;
typedef __attribute__((ext_vector_type(8)))  float  v8f;

union Frag { v16bf v; uint4 u[2]; };

__device__ __forceinline__ unsigned short f32_to_bf16_rne(float f) {
    unsigned u = __builtin_bit_cast(unsigned, f);
    unsigned r = (u + 0x7FFFu + ((u >> 16) & 1u)) >> 16;
    return (unsigned short)r;
}

// ---------------- degree: one atomic per edge ----------------
__global__ void sage_deg_kernel(const int* __restrict__ ei, float* __restrict__ deg, int E_) {
    int e = blockIdx.x * blockDim.x + threadIdx.x;
    if (e < E_) atomicAdd(&deg[ei[E_ + e]], 1.0f);
}

// ---------------- scatter-add of source features at dst ----------------
// 32 threads per edge, 4 features each -> coalesced global_atomic_add_f32
__global__ void sage_scatter_kernel(const float* __restrict__ x, const int* __restrict__ ei,
                                    float* __restrict__ agg, int E_) {
    long t = (long)blockIdx.x * blockDim.x + threadIdx.x;
    long e = t >> 5;
    if (e >= E_) return;
    int f = ((int)t & 31) * 4;
    int s = ei[e];
    int d = ei[E_ + e];
    const float* xs = x + (long)s * DH + f;
    float* ad = agg + (long)d * DH + f;
    atomicAdd(&ad[0], xs[0]);
    atomicAdd(&ad[1], xs[1]);
    atomicAdd(&ad[2], xs[2]);
    atomicAdd(&ad[3], xs[3]);
}

// ---------------- f32 -> bf16 weight conversion ----------------
__global__ void sage_cvt_kernel(const float* __restrict__ src, unsigned short* __restrict__ dst, int n) {
    int i = blockIdx.x * blockDim.x + threadIdx.x;
    if (i < n) dst[i] = f32_to_bf16_rne(src[i]);
}

// ---------------- fused: mean + dual GEMM (mean*Wl^T + x*Wr^T) + bias + relu ----------------
// block = 256 threads = 8 waves; block handles 16 rows; wave w handles output cols [16w,16w+16)
__global__ __launch_bounds__(256) void sage_gemm_kernel(
    const float* __restrict__ x,            // [N,128] root features
    const float* __restrict__ agg,          // [N,128] neighbor feature sums
    const float* __restrict__ deg,          // [N]
    const unsigned short* __restrict__ Wl,  // [128,128] bf16 (row-major: row n = output col)
    const unsigned short* __restrict__ Wr,  // [128,128] bf16
    const float* __restrict__ bl,           // [128]
    float* __restrict__ out,                // [N,128]
    int relu)
{
    __shared__ __align__(16) unsigned short sm_mean[16 * DH];
    __shared__ __align__(16) unsigned short sm_x[16 * DH];

    const int row0 = blockIdx.x * 16;
    const int tid  = threadIdx.x;

    // Stage 16x128 tiles of mean (agg/deg) and x into LDS as bf16.
    {
        int r = tid >> 4;              // 0..15
        int c = (tid & 15) * 8;        // 0..120
        long g = (long)(row0 + r) * DH + c;
        float dinv = 1.0f / fmaxf(deg[row0 + r], 1.0f);
        #pragma unroll
        for (int k = 0; k < 8; ++k) {
            sm_mean[r * DH + c + k] = f32_to_bf16_rne(agg[g + k] * dinv);
            sm_x[r * DH + c + k]    = f32_to_bf16_rne(x[g + k]);
        }
    }
    __syncthreads();

    const int wave = tid >> 5;         // output column tile
    const int lane = tid & 31;
    const int half = lane >> 4;        // 0 or 1
    const int n16  = lane & 15;
    const int ncol = wave * 16 + n16;  // global output column for this lane

    v8f c;
    float bias = bl[ncol];
    #pragma unroll
    for (int r = 0; r < 8; ++r) c[r] = bias;

    const uint4* wl4  = (const uint4*)(Wl + ncol * DH);  // 8 bf16 per uint4
    const uint4* wr4  = (const uint4*)(Wr + ncol * DH);
    const uint4* smm4 = (const uint4*)sm_mean;
    const uint4* smx4 = (const uint4*)sm_x;

    #pragma unroll
    for (int kt = 0; kt < 4; ++kt) {
        const int k0 = kt * 32;
        // A-fragment (16x32 bf16): row = n16; K = {k0+half*8 .. +7} then {k0+16+half*8 .. +7}
        const int a0 = (n16 * DH + k0 + half * 8) >> 3;
        const int a1 = (n16 * DH + k0 + 16 + half * 8) >> 3;
        // B-fragment (32x16 bf16): col = n16 (row ncol of W); K = {k0+half*8 .. +7} then {k0+16+half*8 .. +7}
        const int b0 = (k0 + half * 8) >> 3;

        Frag am, ax, bwl, bwr;
        am.u[0]  = smm4[a0]; am.u[1]  = smm4[a1];
        ax.u[0]  = smx4[a0]; ax.u[1]  = smx4[a1];
        bwl.u[0] = wl4[b0];  bwl.u[1] = wl4[b0 + 2];
        bwr.u[0] = wr4[b0];  bwr.u[1] = wr4[b0 + 2];

        c = __builtin_amdgcn_wmma_f32_16x16x32_bf16(false, am.v, false, bwl.v, (short)0, c, false, false);
        c = __builtin_amdgcn_wmma_f32_16x16x32_bf16(false, ax.v, false, bwr.v, (short)0, c, false, false);
    }

    // D layout: VGPR r -> row (half*8 + r), col n16
    #pragma unroll
    for (int r = 0; r < 8; ++r) {
        float v = c[r];
        if (relu) v = fmaxf(v, 0.0f);
        out[(long)(row0 + half * 8 + r) * DH + ncol] = v;
    }
}

extern "C" void kernel_launch(void* const* d_in, const int* in_sizes, int n_in,
                              void* d_out, int out_size, void* d_ws, size_t ws_size,
                              hipStream_t stream) {
    const float* x   = (const float*)d_in[0];
    const int*   ei  = (const int*)d_in[1];
    const float* Wl0 = (const float*)d_in[2];
    const float* bl0 = (const float*)d_in[3];
    const float* Wr0 = (const float*)d_in[4];
    const float* Wl1 = (const float*)d_in[5];
    const float* bl1 = (const float*)d_in[6];
    const float* Wr1 = (const float*)d_in[7];
    const float* Wl2 = (const float*)d_in[8];
    const float* bl2 = (const float*)d_in[9];
    const float* Wr2 = (const float*)d_in[10];

    const int N_ = in_sizes[0] / DH;       // 100000
    const int E_ = in_sizes[1] / 2;        // 1600000

    // workspace layout
    char* ws = (char*)d_ws;
    const size_t feat_bytes = (size_t)N_ * DH * sizeof(float);
    float*          agg  = (float*)ws;                      ws += feat_bytes;
    float*          hA   = (float*)ws;                      ws += feat_bytes;
    float*          hB   = (float*)ws;                      ws += feat_bytes;
    float*          deg  = (float*)ws;                      ws += (size_t)N_ * sizeof(float);
    unsigned short* wbf  = (unsigned short*)ws;             // 6 * 128*128 bf16
    const int WN = DH * DH;
    unsigned short* Wl0b = wbf + 0 * WN;
    unsigned short* Wr0b = wbf + 1 * WN;
    unsigned short* Wl1b = wbf + 2 * WN;
    unsigned short* Wr1b = wbf + 3 * WN;
    unsigned short* Wl2b = wbf + 4 * WN;
    unsigned short* Wr2b = wbf + 5 * WN;

    // degree (edge-structure invariant across layers: compute once)
    hipMemsetAsync(deg, 0, (size_t)N_ * sizeof(float), stream);
    sage_deg_kernel<<<(E_ + 255) / 256, 256, 0, stream>>>(ei, deg, E_);

    // convert all weights to bf16
    const int cvtBlocks = (WN + 255) / 256;
    sage_cvt_kernel<<<cvtBlocks, 256, 0, stream>>>(Wl0, Wl0b, WN);
    sage_cvt_kernel<<<cvtBlocks, 256, 0, stream>>>(Wr0, Wr0b, WN);
    sage_cvt_kernel<<<cvtBlocks, 256, 0, stream>>>(Wl1, Wl1b, WN);
    sage_cvt_kernel<<<cvtBlocks, 256, 0, stream>>>(Wr1, Wr1b, WN);
    sage_cvt_kernel<<<cvtBlocks, 256, 0, stream>>>(Wl2, Wl2b, WN);
    sage_cvt_kernel<<<cvtBlocks, 256, 0, stream>>>(Wr2, Wr2b, WN);

    const long scatterThreads = (long)E_ * 32;
    const int scatterBlocks = (int)((scatterThreads + 255) / 256);
    const int gemmBlocks = N_ / 16;   // N = 100000 = 6250 * 16 exactly

    // ---- layer 0: x -> hA (relu) ----
    hipMemsetAsync(agg, 0, feat_bytes, stream);
    sage_scatter_kernel<<<scatterBlocks, 256, 0, stream>>>(x, ei, agg, E_);
    sage_gemm_kernel<<<gemmBlocks, 256, 0, stream>>>(x, agg, deg, Wl0b, Wr0b, bl0, hA, 1);

    // ---- layer 1: hA -> hB (relu) ----
    hipMemsetAsync(agg, 0, feat_bytes, stream);
    sage_scatter_kernel<<<scatterBlocks, 256, 0, stream>>>(hA, ei, agg, E_);
    sage_gemm_kernel<<<gemmBlocks, 256, 0, stream>>>(hA, agg, deg, Wl1b, Wr1b, bl1, hB, 1);

    // ---- layer 2: hB -> d_out (no relu) ----
    hipMemsetAsync(agg, 0, feat_bytes, stream);
    sage_scatter_kernel<<<scatterBlocks, 256, 0, stream>>>(hB, ei, agg, E_);
    sage_gemm_kernel<<<gemmBlocks, 256, 0, stream>>>(hB, agg, deg, Wl2b, Wr2b, bl2, (float*)d_out, 0);
}
